// interact_4380866642182
// MI455X (gfx1250) — matrix operand order repaired and checked by reading, not compiled, not run on hardware
//
#include <hip/hip_runtime.h>

typedef __attribute__((ext_vector_type(16))) _Float16     v16h;
typedef __attribute__((ext_vector_type(8)))  float        v8f;
typedef __attribute__((ext_vector_type(8)))  unsigned int v8u;

#define BATCH 16
#define CH    64
#define H     256
#define W     256
#define PH    128
#define PW    128

// ---------------------------------------------------------------------------
// Kernel 1: 3x3 avg-pool, stride 2, pad 1, count_include_pad (always /9).
// One 128-thread block computes one pooled row (128 outputs) of one (b,c).
// Feature row loads are fully coalesced float2; window overlap resolved in LDS.
// ---------------------------------------------------------------------------
__global__ __launch_bounds__(128) void pool3x3s2_kernel(
    const float* __restrict__ feat, float* __restrict__ pooled) {
  const int py = blockIdx.x & (PH - 1);
  const int bc = blockIdx.x >> 7;          // b*64 + c
  const int t  = threadIdx.x;              // 0..127 -> pooled col
  __shared__ float rows[3][264];           // cols -1..256 at idx 0..257 (padded)

  const float* src = feat + (size_t)bc * (H * W);
  const int fy0 = 2 * py - 1;
#pragma unroll
  for (int i = 0; i < 3; ++i) {
    const int fy = fy0 + i;                // -1..255 (255 max since py<=127)
    float2 v;
    if (fy >= 0) {
      v = *(const float2*)(src + fy * W + 2 * t);
    } else {
      v.x = 0.f; v.y = 0.f;                // top zero-pad row
    }
    rows[i][1 + 2 * t] = v.x;
    rows[i][2 + 2 * t] = v.y;
  }
  if (t < 3) { rows[t][0] = 0.f; rows[t][W + 1] = 0.f; }  // left/right pad cols
  __syncthreads();

  float s = 0.f;
#pragma unroll
  for (int i = 0; i < 3; ++i)
    s += rows[i][2 * t] + rows[i][2 * t + 1] + rows[i][2 * t + 2];
  pooled[(size_t)bc * (PH * PW) + py * PW + t] = s * (1.0f / 9.0f);
}

// ---------------------------------------------------------------------------
// Kernel 2: per-pixel channel reduction over 8 shifts via WMMA diagonal trick.
//   wave = 16 pixels (one x-run of an output row). lane: m = lane&15 (pixel),
//   h = lane>>4 (channel half: channels 32h..32h+31 live in this lane's regs).
// Per shift: diff f16 pairs staged to LDS as tile[c2][m] (c2 = channel/2),
// then loaded in the exact A(16x32) / B(32x16) f16 per-lane layouts
// (CDNA5 ISA 7.12.2): every A/B VGPR is one ds_load_b32.
//   WMMA#0 uses channels 0..31 (c2 0..15), WMMA#1 channels 32..63 (c2 16..31).
//   A: lane h*16+L, vgpr j: c2 = (j<4) ? h*4+j : 8+h*4+(j-4)   (M=L)
//   B: lane h*16+n, vgpr j: c2 = (j<4) ? h*4+j : 8+h*4+(j-4)   (N=n)
// D[m][n] = sum_k diff[k][m]*diff[k][n]; diagonal = sum of squares.
// Running max kept in C-layout; diagonal extracted once via ds_bpermute:
//   D[m][m]: m<8 -> vgpr m, lane m;  m>=8 -> vgpr m-8, lane 16+m.
// ---------------------------------------------------------------------------
__global__ __launch_bounds__(256) void shiftdiff_max_kernel(
    const float* __restrict__ feat, const float* __restrict__ pooled,
    const int* __restrict__ distp, float* __restrict__ out) {
  const int bid  = blockIdx.x;             // BATCH * H * 2 blocks
  const int xt   = bid & 1;
  const int y    = (bid >> 1) & (H - 1);
  const int b    = bid >> 9;
  const int wv   = threadIdx.x >> 5;       // wave in block (0..7)
  const int lane = threadIdx.x & 31;
  const int m    = lane & 15;              // pixel within 16-wide tile
  const int h    = lane >> 4;              // channel half
  const int x0   = xt * 128 + wv * 16;
  const int d    = 2 * distp[0];           // dist=1 -> d=2

  __shared__ unsigned int tile[8 * 544];   // 8 waves * 32 rows * stride 17
  unsigned int* tl = tile + wv * 544;

  // Load this lane's 32 feature channel values (channels 32h+2j, 32h+2j+1).
  const int fbase = ((b * CH + 32 * h) * H + y) * W + (x0 + m);
  float f0[16], f1[16];
#pragma unroll
  for (int j = 0; j < 16; ++j) {
    f0[j] = feat[fbase + (2 * j)     * (H * W)];
    f1[j] = feat[fbase + (2 * j + 1) * (H * W)];
  }

  const int dy[8] = { d, d,  d, 0,  0, -d, -d, -d };
  const int dx[8] = { d, 0, -d, d, -d,  d,  0, -d };

  v8f maxC = {0.f, 0.f, 0.f, 0.f, 0.f, 0.f, 0.f, 0.f};  // sums >= 0

#pragma unroll
  for (int s = 0; s < 8; ++s) {
    // roll(up, (sy,sx)): up index = (y-sy)&255, (x-sx)&255; up = pooled[>>1]
    const int r     = ((y - dy[s]) & (H - 1)) >> 1;
    const int c     = ((x0 + m - dx[s]) & (W - 1)) >> 1;
    const int pbase = ((b * CH + 32 * h) * PH + r) * PW + c;
#pragma unroll
    for (int j = 0; j < 16; ++j) {
      const float u0 = pooled[pbase + (2 * j)     * (PH * PW)];
      const float u1 = pooled[pbase + (2 * j + 1) * (PH * PW)];
      const auto pk  = __builtin_amdgcn_cvt_pkrtz(f0[j] - u0, f1[j] - u1);
      tl[(16 * h + j) * 17 + m] = __builtin_bit_cast(unsigned int, pk);
    }
    __syncthreads();  // order LDS tile stores before layout loads

    v8u au0, au1, bu0, bu1;
#pragma unroll
    for (int j = 0; j < 8; ++j) {
      const int c2a = (j < 4) ? (h * 4 + j) : (8 + h * 4 + (j - 4));
      au0[j] = tl[c2a * 17 + m];
      au1[j] = tl[(16 + c2a) * 17 + m];
      const int c2b = (j < 4) ? (h * 4 + j) : (8 + h * 4 + (j - 4));
      bu0[j] = tl[c2b * 17 + m];
      bu1[j] = tl[(16 + c2b) * 17 + m];
    }
    v8f acc = {0.f, 0.f, 0.f, 0.f, 0.f, 0.f, 0.f, 0.f};
    acc = __builtin_amdgcn_wmma_f32_16x16x32_f16(
        false, __builtin_bit_cast(v16h, au0),
        false, __builtin_bit_cast(v16h, bu0), (short)0, acc, false, false);
    acc = __builtin_amdgcn_wmma_f32_16x16x32_f16(
        false, __builtin_bit_cast(v16h, au1),
        false, __builtin_bit_cast(v16h, bu1), (short)0, acc, false, false);
#pragma unroll
    for (int r8 = 0; r8 < 8; ++r8) maxC[r8] = fmaxf(maxC[r8], acc[r8]);
    __syncthreads();  // tile reused next shift
  }

  // Diagonal extraction: lane wants D[m][m] from lane (m<8 ? m : 16+m), vgpr m&7.
  const int srcByte = ((m < 8) ? m : (m + 16)) * 4;
  float t8[8];
#pragma unroll
  for (int r8 = 0; r8 < 8; ++r8)
    t8[r8] = __builtin_bit_cast(
        float, __builtin_amdgcn_ds_bpermute(
                   srcByte, __builtin_bit_cast(int, maxC[r8])));
  const int sel = m & 7;
  float v = t8[0];
#pragma unroll
  for (int r8 = 1; r8 < 8; ++r8)
    if (sel == r8) v = t8[r8];
  if (h == 0) out[(b * H + y) * W + x0 + m] = v;
}

// ---------------------------------------------------------------------------
extern "C" void kernel_launch(void* const* d_in, const int* in_sizes, int n_in,
                              void* d_out, int out_size, void* d_ws, size_t ws_size,
                              hipStream_t stream) {
  (void)in_sizes; (void)n_in; (void)out_size; (void)ws_size;
  const float* feat  = (const float*)d_in[0];
  const int*   dist  = (const int*)d_in[1];
  float*       pooled = (float*)d_ws;          // 16*64*128*128 f32 = 64 MB
  float*       out    = (float*)d_out;

  // Pass A: pooled = avg_pool3x3(stride2, pad1) of feature
  pool3x3s2_kernel<<<BATCH * CH * PH, 128, 0, stream>>>(feat, pooled);
  // Pass B: fused (feature - rolled upsample)^2 channel-sum + max over shifts
  shiftdiff_max_kernel<<<BATCH * H * 2, 256, 0, stream>>>(feat, pooled, dist, out);
}
